// CRFLayer_46660524703784
// MI455X (gfx1250) — matrix operand order repaired and checked by reading, not compiled, not run on hardware
//
#include <hip/hip_runtime.h>

#define SS 512
#define BB 512
#define TT 64
#define NEGV (-10000.0f)
#define START_TAG 62
#define STOP_TAG 63

typedef __attribute__((ext_vector_type(16))) _Float16 v16h;
typedef __attribute__((ext_vector_type(8)))  float    v8f;

// ---------------------------------------------------------------------------
// Forward algorithm: one wave (32 lanes) owns 16 batch rows. Per step:
//   P = exp(la - rowmax) @ exp(trans)   via v_wmma_f32_16x16x32_f16
//   la' = log(P) + rowmax + feats[t]    (masked)
// exp(trans) B-fragments live in VGPRs (step-invariant). feats tiles are
// double-buffered into LDS with global_load_async_to_lds_b128 (ASYNCcnt),
// overlapping the HBM fetch for step t+1 with step t's compute.
// ---------------------------------------------------------------------------
__global__ __launch_bounds__(32)
void crf_forward_kernel(const float* __restrict__ feats,
                        const float* __restrict__ mask,
                        const float* __restrict__ trans,
                        float* __restrict__ log_z)
{
    __shared__ float la[16][72];        // padded stride vs 64-bank LDS
    __shared__ float rowmax[16];
    __shared__ float fbuf[2][16][64];   // double-buffered feats tile (2 x 4KB)

    const int lane = threadIdx.x;   // 0..31
    const int grp  = lane >> 4;     // 0/1 (lane half)
    const int nlo  = lane & 15;
    const int b0   = blockIdx.x * 16;

    // init log_alpha: NEG everywhere except START column
    for (int idx = lane; idx < 16 * TT; idx += 32) {
        int m = idx >> 6, t = idx & 63;
        la[m][t] = (t == START_TAG) ? 0.0f : NEGV;
    }

    // Async-copy one 4KB feats tile (16 rows x 64 f32, contiguous) into LDS.
    // 8 wave-wide b128 issues; tracked with ASYNCcnt.
    auto issue_tile = [&](int t, int nb) {
        unsigned lbase = (unsigned)(uintptr_t)(&fbuf[nb][0][0]) + (unsigned)lane * 16u;
        unsigned long long gbase =
            (unsigned long long)(feats + ((size_t)t * BB + b0) * TT) + (unsigned long long)lane * 16ull;
        #pragma unroll
        for (int c = 0; c < 8; ++c) {
            unsigned l = lbase + (unsigned)c * 512u;
            unsigned long long g = gbase + (unsigned long long)c * 512ull;
            asm volatile("global_load_async_to_lds_b128 %0, %1, off"
                         :: "v"(l), "v"(g) : "memory");
        }
    };

    // B-matrix fragments: E[k][n] = exp(trans[k][n]), f16, 2 K-frags x 4 N-tiles.
    // B k order used here (same as A): lanes 0-15 hold K=0..15 (2 per VGPR), lanes 16-31 K=16..31.
    v16h Bfrag[2][4];
    #pragma unroll
    for (int c = 0; c < 2; ++c) {
        #pragma unroll
        for (int j = 0; j < 4; ++j) {
            #pragma unroll
            for (int e = 0; e < 16; ++e) {
                int k = 32 * c + 16 * grp + e;
                int n = 16 * j + nlo;
                Bfrag[c][j][e] = (_Float16)__expf(trans[k * TT + n]);
            }
        }
    }

    issue_tile(0, 0);                 // prefetch tile for t=0
    __syncthreads();

    for (int t = 0; t < SS; ++t) {
        const int cur = t & 1;
        if (t + 1 < SS) issue_tile(t + 1, (t + 1) & 1);   // overlap next fetch

        // per-row max (all 32 lanes: each reduces half a row, combine via shfl)
        {
            float pm = la[nlo][32 * grp];
            #pragma unroll 8
            for (int k = 1; k < 32; ++k) pm = fmaxf(pm, la[nlo][32 * grp + k]);
            pm = fmaxf(pm, __shfl_xor(pm, 16));
            if (grp == 0) rowmax[nlo] = pm;
        }
        __syncthreads();

        // A-matrix fragments: exp(la - rowmax). k order used here (same as B):
        // lane L -> row M=L&15, k = 16*(L>>4) + e
        const float mb = rowmax[nlo];
        v16h Afrag[2];
        #pragma unroll
        for (int c = 0; c < 2; ++c) {
            #pragma unroll
            for (int e = 0; e < 16; ++e) {
                int k = 32 * c + 16 * grp + e;
                Afrag[c][e] = (_Float16)__expf(la[nlo][k] - mb);
            }
        }

        // GEMM: P[16x64] = A[16x64] @ E[64x64], 8 WMMAs
        v8f C[4];
        #pragma unroll
        for (int j = 0; j < 4; ++j) {
            v8f acc = {};
            acc = __builtin_amdgcn_wmma_f32_16x16x32_f16(
                false, Afrag[0], false, Bfrag[0][j], (short)0, acc, false, false);
            acc = __builtin_amdgcn_wmma_f32_16x16x32_f16(
                false, Afrag[1], false, Bfrag[1][j], (short)0, acc, false, false);
            C[j] = acc;
        }
        __syncthreads();   // all lanes done reading la before overwrite

        // make sure tile t is fully in LDS (tile t+1 may still be in flight)
        if (t + 1 < SS) asm volatile("s_wait_asynccnt 0x8" ::: "memory");
        else            asm volatile("s_wait_asynccnt 0x0" ::: "memory");

        // la' = log(P) + rowmax + feats[t], where mask > 0
        const size_t mbase = (size_t)t * BB + b0;
        #pragma unroll
        for (int j = 0; j < 4; ++j) {
            #pragma unroll
            for (int r = 0; r < 8; ++r) {
                int m = r + 8 * grp;       // D layout: M = r + 8*(lane>>4)
                int n = 16 * j + nlo;      //           N = lane & 15
                float p  = C[j][r];
                float lp = (p > 0.0f) ? __logf(p) : NEGV;
                float nv = lp + rowmax[m] + fbuf[cur][m][n];
                if (mask[mbase + m] > 0.0f) la[m][n] = nv;
            }
        }
        __syncthreads();
    }

    // log_z[b] = logsumexp(la[b,:] + trans[STOP,:])
    if (lane < 16) {
        float mx = -3.4e38f;
        #pragma unroll 8
        for (int n = 0; n < TT; ++n)
            mx = fmaxf(mx, la[lane][n] + trans[STOP_TAG * TT + n]);
        float s = 0.0f;
        #pragma unroll 8
        for (int n = 0; n < TT; ++n)
            s += __expf(la[lane][n] + trans[STOP_TAG * TT + n] - mx);
        log_z[b0 + lane] = mx + __logf(s);
    }
}

// ---------------------------------------------------------------------------
// Gold-path score: one thread per batch element.
// ---------------------------------------------------------------------------
__global__ __launch_bounds__(256)
void crf_score_kernel(const float* __restrict__ feats,
                      const int*   __restrict__ tags,
                      const float* __restrict__ mask,
                      const float* __restrict__ trans,
                      float* __restrict__ score)
{
    int b = blockIdx.x * blockDim.x + threadIdx.x;
    if (b >= BB) return;
    float acc = 0.0f, msum = 0.0f;
    int prev = START_TAG;
    for (int s = 0; s < SS; ++s) {
        int   tg = tags[(size_t)s * BB + b];
        float mk = mask[(size_t)s * BB + b];
        float em = feats[((size_t)s * BB + b) * TT + tg];
        float tr = trans[prev * TT + tg];
        acc  += (em + tr) * mk;
        msum += mk;
        prev  = tg;
    }
    int last_idx = (int)msum - 1;
    if (last_idx < 0) last_idx = 0;
    if (last_idx >= SS) last_idx = SS - 1;
    int last = tags[(size_t)last_idx * BB + b];
    score[b] = acc + trans[last * TT + STOP_TAG];
}

// ---------------------------------------------------------------------------
// Final reduction: mean(log_z - score)
// ---------------------------------------------------------------------------
__global__ __launch_bounds__(256)
void crf_reduce_kernel(const float* __restrict__ log_z,
                       const float* __restrict__ score,
                       float* __restrict__ out)
{
    __shared__ float buf[256];
    int tid = threadIdx.x;
    float s = 0.0f;
    for (int b = tid; b < BB; b += 256) s += log_z[b] - score[b];
    buf[tid] = s;
    __syncthreads();
    for (int w = 128; w > 0; w >>= 1) {
        if (tid < w) buf[tid] += buf[tid + w];
        __syncthreads();
    }
    if (tid == 0) out[0] = buf[0] / (float)BB;
}

extern "C" void kernel_launch(void* const* d_in, const int* in_sizes, int n_in,
                              void* d_out, int out_size, void* d_ws, size_t ws_size,
                              hipStream_t stream) {
    const float* feats = (const float*)d_in[0];   // (S,B,T) f32
    const int*   tags  = (const int*)  d_in[1];   // (S,B)   i32
    const float* mask  = (const float*)d_in[2];   // (S,B)   f32
    const float* trans = (const float*)d_in[3];   // (T,T)   f32

    float* log_z = (float*)d_ws;        // [BB]
    float* score = log_z + BB;          // [BB]

    crf_forward_kernel<<<BB / 16, 32, 0, stream>>>(feats, mask, trans, log_z);
    crf_score_kernel<<<BB / 256, 256, 0, stream>>>(feats, tags, mask, trans, score);
    crf_reduce_kernel<<<1, 256, 0, stream>>>(log_z, score, (float*)d_out);
}
